// CNNLSTM_7773890805997
// MI455X (gfx1250) — matrix operand order repaired and checked by reading, not compiled, not run on hardware
//
#include <hip/hip_runtime.h>
#include <cstddef>

// ---------------------------------------------------------------------------
// CDNA5 (gfx1250) wave32 WMMA types & helpers
// ---------------------------------------------------------------------------
typedef __attribute__((ext_vector_type(16))) __bf16 v16bf;
typedef __attribute__((ext_vector_type(8)))  float  v8f;

#define GAS __attribute__((address_space(1)))

__device__ __forceinline__ unsigned short f2bf_bits(float f) {
  unsigned u = __builtin_bit_cast(unsigned, f);
  u += 0x7fffu + ((u >> 16) & 1u);           // round-to-nearest-even
  return (unsigned short)(u >> 16);
}
__device__ __forceinline__ float bf_bits2f(unsigned short h) {
  unsigned u = ((unsigned)h) << 16;
  return __builtin_bit_cast(float, u);
}
__device__ __forceinline__ __bf16 bits2bf(unsigned short h) {
  return __builtin_bit_cast(__bf16, h);
}

// Force global (AS1) vector memory ops: avoids flat_load_* which ticks both
// LOADcnt and DScnt and shares the LDS pipe.
__device__ __forceinline__ v16bf ldg_v16bf(const unsigned short* p) {
  return *(const GAS v16bf*)p;
}
__device__ __forceinline__ unsigned short ldg_u16(const unsigned short* p) {
  return *(const GAS unsigned short*)p;
}
__device__ __forceinline__ float ldg_f32(const float* p) {
  return *(const GAS float*)p;
}
__device__ __forceinline__ void stg_u16(unsigned short* p, unsigned short v) {
  *(GAS unsigned short*)p = v;
}
__device__ __forceinline__ void stg_f32(float* p, float v) {
  *(GAS float*)p = v;
}

__device__ __forceinline__ unsigned short ld_bits(const float* p) { return f2bf_bits(ldg_f32(p)); }
__device__ __forceinline__ unsigned short ld_bits(const unsigned short* p) { return ldg_u16(p); }

// Opaque pointer "launder": redefines the SSA value each call so loads through
// it cannot be hoisted out of (or CSE'd across) the surrounding loop, while
// remaining ordinary cached RT-scope loads the scheduler can batch/overlap.
#define LAUNDER_PTR(p) asm volatile("" : "+s"(p))

// Fast transcendentals: gfx1250 has V_TANH_F32 in hardware.
__device__ __forceinline__ float fast_exp(float x) {
#if __has_builtin(__builtin_amdgcn_exp2f)
  return __builtin_amdgcn_exp2f(x * 1.4426950408889634f);
#else
  return __expf(x);
#endif
}
__device__ __forceinline__ float fast_rcp(float x) {
#if __has_builtin(__builtin_amdgcn_rcpf)
  return __builtin_amdgcn_rcpf(x);
#else
  return 1.0f / x;
#endif
}
__device__ __forceinline__ float fast_tanh(float x) {
#if __has_builtin(__builtin_amdgcn_tanhf)
  return __builtin_amdgcn_tanhf(x);
#elif __has_builtin(__builtin_amdgcn_tanh_f32)
  return __builtin_amdgcn_tanh_f32(x);
#else
  float e = fast_exp(-2.0f * x);
  return (1.0f - e) * fast_rcp(1.0f + e);
#endif
}
__device__ __forceinline__ float fast_sigm(float x) {
  return 0.5f * fast_tanh(0.5f * x) + 0.5f;   // sigma(x) = (1+tanh(x/2))/2
}

__device__ __forceinline__ v8f wmma_bf16(v16bf a, v16bf b, v8f c) {
  // D = A(16x32 bf16) * B(32x16 bf16) + C(16x16 f32)
  return __builtin_amdgcn_wmma_f32_16x16x32_bf16(false, a, false, b, (short)0, c,
                                                 false, false);
}

// A-fragment K index for 16-bit A (16x32), contiguous order:
//   lanes 0-15 : K 0..15; lanes 16-31: K 16..31
__device__ __forceinline__ int a_klocal(int e, bool hi) {
  return e + (hi ? 16 : 0);
}

// ---------------------------------------------------------------------------
// Weight packing kernels (run once per launch, tiny)
// ---------------------------------------------------------------------------
__global__ void pack_conv_kernel(const float* __restrict__ w, unsigned short* __restrict__ dst,
                                 int CO, int CIK, int nchunks, int mtiles)
{
  int total = mtiles * nchunks * 512;
  for (int idx = blockIdx.x * blockDim.x + threadIdx.x; idx < total;
       idx += gridDim.x * blockDim.x) {
    int frag = idx >> 9;
    int r    = idx & 511;
    int lane = r >> 4;
    int e    = r & 15;
    int mtile = frag / nchunks;
    int c     = frag - mtile * nchunks;
    int m  = mtile * 16 + (lane & 15);
    int kl = a_klocal(e, lane >= 16);
    int kk = 32 * c + kl;
    float v = 0.0f;
    if (m < CO && kk < CIK) v = ldg_f32(w + (size_t)m * CIK + kk);
    stg_u16(dst + idx, f2bf_bits(v));
  }
}

// Pack B = [Wih | Whh]^T into B-fragment order: lane n<16 holds col n, K rows
// 0..15 contiguous; lanes 16-31 hold K rows 16..31.
__global__ void pack_lstm_kernel(const float* __restrict__ Wih, const float* __restrict__ Whh,
                                 unsigned short* __restrict__ dst, int IN, int H, int NT, int KC)
{
  int total = NT * KC * 512;
  for (int idx = blockIdx.x * blockDim.x + threadIdx.x; idx < total;
       idx += gridDim.x * blockDim.x) {
    int frag = idx >> 9;
    int r    = idx & 511;
    int lane = r >> 4;
    int e    = r & 15;
    int nt = frag / KC;
    int kc = frag - nt * KC;
    int gate = nt * 16 + (lane & 15);
    int kl = 32 * kc + e + ((lane >= 16) ? 16 : 0);
    float v = 0.0f;
    if (kl < IN)           v = ldg_f32(Wih + (size_t)gate * IN + kl);
    else if (kl < IN + H)  v = ldg_f32(Whh + (size_t)gate * H + (kl - IN));
    stg_u16(dst + idx, f2bf_bits(v));
  }
}

__global__ void bias_sum_kernel(const float* __restrict__ a, const float* __restrict__ b,
                                float* __restrict__ dst, int n)
{
  int i = blockIdx.x * blockDim.x + threadIdx.x;
  if (i < n) stg_f32(dst + i, ldg_f32(a + i) + ldg_f32(b + i));
}

// ---------------------------------------------------------------------------
// Conv1d (VALID) + bias + ReLU + maxpool2 as implicit GEMM with bf16 WMMA.
// Block = 64 threads (2 waves). Wave w computes output-channel tile w (16 rows)
// for one (batch, 16-conv-position tile). D cols = conv positions; pool pairs
// adjacent columns via a lane shuffle. im2col gather walks (channel, offset)
// incrementally -- no per-element integer division.
// ---------------------------------------------------------------------------
template <typename TIN>
__global__ __launch_bounds__(64) void conv_pool_kernel(
    const TIN* __restrict__ in, const unsigned short* __restrict__ wpack,
    const float* __restrict__ bias, unsigned short* __restrict__ out,
    int CI, int CO, int K, int Lin, int Lc, int Lp, int nchunks, int mtiles)
{
  __shared__ unsigned short xlds[1600];  // max CI*(K+16) over all layers = 1540
  const int b    = blockIdx.y;
  const int l0   = blockIdx.x * 16;
  const int tid  = threadIdx.x;
  const int wave = tid >> 5;
  const int lane = tid & 31;
  const int XS   = K + 16;

  // Stage input window x[b][i][l0 .. l0+15+K-1] into LDS as bf16.
  for (int idx = tid; idx < CI * (K + 15); idx += 64) {
    int i = idx / (K + 15);
    int j = idx - i * (K + 15);
    int g = l0 + j;
    unsigned short bits = 0;
    if (g < Lin) bits = ld_bits(in + ((size_t)b * CI + i) * (size_t)Lin + g);
    xlds[i * XS + j] = bits;
  }
  __syncthreads();

  const int mtile = wave;
  if (mtile >= mtiles) return;  // wave-uniform
  const int  nn = lane & 15;
  const bool hi = lane >= 16;
  const int  CIK = CI * K;

  // incremental (channel, k-offset) walker for this lane's first element
  int ich = 0;
  int ko  = hi ? 16 : 0;
  while (ko >= K) { ko -= K; ++ich; }

  v8f acc = {};
  for (int c = 0; c < nchunks; ++c) {
    // A: pre-packed weights, contiguous 32B per lane
    v16bf a = ldg_v16bf(wpack + ((size_t)(mtile * nchunks + c) << 9) + lane * 16);
    // B: im2col gather from LDS, B[kk][n] = x[i][n + k], kk = i*K + k
    int lim = CIK - (32 * c + (hi ? 16 : 0));  // valid elements this half-chunk
    v16bf bm;
    int ii = ich, kk2 = ko;
#pragma unroll
    for (int e = 0; e < 16; ++e) {
      unsigned short bits = 0;
      if (e < lim) bits = xlds[ii * XS + nn + kk2];
      bm[e] = bits2bf(bits);
      ++kk2;
      if (kk2 == K) { kk2 = 0; ++ii; }
    }
    acc = wmma_bf16(a, bm, acc);
    ko += 32;
    while (ko >= K) { ko -= K; ++ich; }
  }

  // bias + relu + maxpool2 + bf16 store
#pragma unroll
  for (int v = 0; v < 8; ++v) {
    int m = mtile * 16 + v + (hi ? 8 : 0);
    float val = acc[v] + ((m < CO) ? ldg_f32(bias + m) : 0.0f);
    val = fmaxf(val, 0.0f);
    float other = __shfl_xor(val, 1, 32);  // partner conv column
    float pm = fmaxf(val, other);
    int col = l0 + nn;
    if (((nn & 1) == 0) && (col + 1 < Lc) && (m < CO)) {
      stg_u16(out + ((size_t)b * CO + m) * (size_t)Lp + (col >> 1), f2bf_bits(pm));
    }
  }
}

// ---------------------------------------------------------------------------
// Fused 2-layer bi-LSTM (T=94) + classifier. One persistent kernel.
// Grid = 8 blocks x 128 threads (4 waves); each wave owns 16 batch rows.
// Per step gates = [x||h] @ [Wih|Whh]^T + (bih+bhh) via WMMA (M=batch,N=gates).
// i/f/g/o tiles stay element-aligned with the c/h fragments -> pure VALU update
// using the hardware V_TANH_F32 path. Weight base pointers are laundered once
// per timestep so fragment loads stay in-loop (L2 hits) without becoming
// 94-iteration live ranges -- and without volatile's SYS-scope serialization.
// ---------------------------------------------------------------------------
__global__ __launch_bounds__(128, 1) void lstm_head_kernel(
    const unsigned short* __restrict__ xin,  // [512][10][94] bf16 bits
    const unsigned short* __restrict__ w1f, const unsigned short* __restrict__ w1r,
    const unsigned short* __restrict__ w2f, const unsigned short* __restrict__ w2r,
    const float* __restrict__ b1f, const float* __restrict__ b1r,
    const float* __restrict__ b2f, const float* __restrict__ b2r,
    const float* __restrict__ clsw, const float* __restrict__ clsb,
    float* __restrict__ outp)
{
  // a1: [64 rows][80 cols] = x(0..9) | h1f(10..41) | h1r(42..73) | zero-pad
  // a2: [64 rows][192 cols] = y1 = h1f(0..31)|h1r(32..63) | h2f(64..127) | h2r(128..191)
  __shared__ unsigned short a1[64 * 80];
  __shared__ unsigned short a2[64 * 192];

  const int tid  = threadIdx.x;
  const int wave = tid >> 5;
  const int lane = tid & 31;
  const int nn   = lane & 15;
  const bool hi  = lane >= 16;
  const int arow = wave * 16 + nn;        // A-matrix row this lane reads
  const int bbase = blockIdx.x * 64;

  const unsigned short* w1p[2] = {w1f, w1r};
  const unsigned short* w2p[2] = {w2f, w2r};

  for (int i = tid; i < 64 * 80;  i += 128) a1[i] = 0;
  for (int i = tid; i < 64 * 192; i += 128) a2[i] = 0;

  v8f zero = {};
  v8f c1[2][2], c2[2][4];
#pragma unroll
  for (int d = 0; d < 2; ++d) {
#pragma unroll
    for (int j = 0; j < 2; ++j) c1[d][j] = zero;
#pragma unroll
    for (int j = 0; j < 4; ++j) c2[d][j] = zero;
  }
  __syncthreads();

  for (int t = 0; t < 94; ++t) {
    // Break cross-iteration invariance of the weight bases (see note above).
    LAUNDER_PTR(w1p[0]);
    LAUNDER_PTR(w1p[1]);
    LAUNDER_PTR(w2p[0]);
    LAUNDER_PTR(w2p[1]);

    // load x_t for 64 batch rows into a1 cols 0..9
    for (int idx = tid; idx < 640; idx += 128) {
      int r  = idx / 10;
      int ch = idx - 10 * r;
      a1[r * 80 + ch] = ldg_u16(xin + ((size_t)(bbase + r) * 10 + ch) * 94 + t);
    }
    __syncthreads();

    // ---------------- layer 1 (dirs f,r), K=42 padded to 64 ----------------
#pragma unroll
    for (int d = 0; d < 2; ++d) {
      const unsigned short* wp = w1p[d];
      const float*          bp = d ? b1r : b1f;
      v16bf af[2];
#pragma unroll
      for (int kc = 0; kc < 2; ++kc) {
#pragma unroll
        for (int e = 0; e < 16; ++e) {
          int kl = 32 * kc + a_klocal(e, hi);
          unsigned short bits = 0;
          if (kl < 10)       bits = a1[arow * 80 + kl];
          else if (kl < 42)  bits = a1[arow * 80 + 10 + 32 * d + (kl - 10)];
          af[kc][e] = bits2bf(bits);
        }
      }
#pragma unroll
      for (int j = 0; j < 2; ++j) {          // hidden-col tiles (h=32 -> 2)
        v8f g4[4];                           // i, f, g, o gate tiles
#pragma unroll
        for (int q = 0; q < 4; ++q) {
          int nt = 2 * q + j;                // gate tile (8 N-tiles total)
          float bv = ldg_f32(bp + nt * 16 + nn);
          v8f acc;
#pragma unroll
          for (int v = 0; v < 8; ++v) acc[v] = bv;
#pragma unroll
          for (int kc = 0; kc < 2; ++kc) {
            v16bf bm = ldg_v16bf(wp + (((size_t)nt * 2 + kc) << 9) + lane * 16);
            acc = wmma_bf16(af[kc], bm, acc);
          }
          g4[q] = acc;
        }
        v8f cn;
#pragma unroll
        for (int v = 0; v < 8; ++v) {
          float ig = fast_sigm(g4[0][v]);
          float fg = fast_sigm(g4[1][v]);
          float gg = fast_tanh(g4[2][v]);
          float og = fast_sigm(g4[3][v]);
          float c  = fg * c1[d][j][v] + ig * gg;
          cn[v] = c;
          float h = og * fast_tanh(c);
          int m = v + (hi ? 8 : 0);
          int r = wave * 16 + m;
          int colh = j * 16 + nn;
          unsigned short hb = f2bf_bits(h);
          a1[r * 80 + 10 + 32 * d + colh] = hb;   // next-step recurrence input
          a2[r * 192 + 32 * d + colh]     = hb;   // y1 for layer 2
        }
        c1[d][j] = cn;
      }
    }
    __syncthreads();

    // ---------------- layer 2 (dirs f,r), K=128 ----------------------------
#pragma unroll
    for (int d = 0; d < 2; ++d) {
      const unsigned short* wp = w2p[d];
      const float*          bp = d ? b2r : b2f;
      v16bf af[4];
#pragma unroll
      for (int kc = 0; kc < 4; ++kc) {
#pragma unroll
        for (int e = 0; e < 16; ++e) {
          int kl  = 32 * kc + a_klocal(e, hi);
          int col = (kl < 64) ? kl : (64 + 64 * d + (kl - 64));
          af[kc][e] = bits2bf(a2[arow * 192 + col]);
        }
      }
#pragma unroll
      for (int j = 0; j < 4; ++j) {          // hidden-col tiles (h=64 -> 4)
        v8f g4[4];
#pragma unroll
        for (int q = 0; q < 4; ++q) {
          int nt = 4 * q + j;                // 16 N-tiles total
          float bv = ldg_f32(bp + nt * 16 + nn);
          v8f acc;
#pragma unroll
          for (int v = 0; v < 8; ++v) acc[v] = bv;
#pragma unroll
          for (int kc = 0; kc < 4; ++kc) {
            v16bf bm = ldg_v16bf(wp + (((size_t)nt * 4 + kc) << 9) + lane * 16);
            acc = wmma_bf16(af[kc], bm, acc);
          }
          g4[q] = acc;
        }
        v8f cn;
#pragma unroll
        for (int v = 0; v < 8; ++v) {
          float ig = fast_sigm(g4[0][v]);
          float fg = fast_sigm(g4[1][v]);
          float gg = fast_tanh(g4[2][v]);
          float og = fast_sigm(g4[3][v]);
          float c  = fg * c2[d][j][v] + ig * gg;
          cn[v] = c;
          float h = og * fast_tanh(c);
          int m = v + (hi ? 8 : 0);
          int r = wave * 16 + m;
          a2[r * 192 + 64 + 64 * d + j * 16 + nn] = f2bf_bits(h);
        }
        c2[d][j] = cn;
      }
    }
    __syncthreads();
  }

  // classifier: out[b] = concat(h2f,h2r) @ cls_w^T + cls_b  (128 -> 2)
  {
    int r  = tid >> 1;
    int cl = tid & 1;
    float s = ldg_f32(clsb + cl);
    for (int jj = 0; jj < 128; ++jj)
      s += bf_bits2f(a2[r * 192 + 64 + jj]) * ldg_f32(clsw + cl * 128 + jj);
    stg_f32(outp + (size_t)(bbase + r) * 2 + cl, s);
  }
}

// ---------------------------------------------------------------------------
// Host launcher
// ---------------------------------------------------------------------------
extern "C" void kernel_launch(void* const* d_in, const int* in_sizes, int n_in,
                              void* d_out, int out_size, void* d_ws, size_t ws_size,
                              hipStream_t stream)
{
  (void)out_size; (void)ws_size;
  static const int CI[6]  = {1, 5, 5, 10, 20, 20};
  static const int CO[6]  = {5, 5, 10, 20, 20, 10};
  static const int KK[6]  = {251, 150, 100, 81, 61, 14};
  static const int LINA[7]= {9000, 4375, 2113, 1007, 463, 201, 94};

  // Input-order detection: "input" (512*1*9000) first, params after; or reversed.
  int input_idx, pbase;
  if (in_sizes[0] == 512 * 9000) { input_idx = 0;        pbase = 1; }
  else                           { input_idx = n_in - 1; pbase = 0; }

  const float* convw[6]; const float* convb[6];
  for (int i = 0; i < 6; ++i) {
    convw[i] = (const float*)d_in[pbase + 2 * i];
    convb[i] = (const float*)d_in[pbase + 2 * i + 1];
  }
  const float *Wih[4], *Whh[4], *bih[4], *bhh[4];
  for (int i = 0; i < 4; ++i) {
    Wih[i] = (const float*)d_in[pbase + 12 + 4 * i + 0];
    Whh[i] = (const float*)d_in[pbase + 12 + 4 * i + 1];
    bih[i] = (const float*)d_in[pbase + 12 + 4 * i + 2];
    bhh[i] = (const float*)d_in[pbase + 12 + 4 * i + 3];
  }
  const float* clsw = (const float*)d_in[pbase + 28];
  const float* clsb = (const float*)d_in[pbase + 29];

  // ---- workspace carve (bytes, 256B aligned) ----
  char* ws = (char*)d_ws;
  size_t off = 0;
  auto alloc = [&](size_t bytes) -> char* {
    char* p = ws + off;
    off = (off + bytes + 255) & ~(size_t)255;
    return p;
  };
  unsigned short* bufA = (unsigned short*)alloc((size_t)512 * 5 * 4375 * 2);
  unsigned short* bufB = (unsigned short*)alloc((size_t)512 * 5 * 2113 * 2);

  unsigned short* cpack[6]; int mt[6], nc[6];
  for (int i = 0; i < 6; ++i) {
    nc[i] = (CI[i] * KK[i] + 31) / 32;
    mt[i] = (CO[i] + 15) / 16;
    cpack[i] = (unsigned short*)alloc((size_t)mt[i] * nc[i] * 512 * 2);
  }
  static const int NT[4]  = {8, 8, 16, 16};
  static const int KC[4]  = {2, 2, 4, 4};
  static const int INS[4] = {10, 10, 64, 64};
  static const int HS[4]  = {32, 32, 64, 64};
  unsigned short* lpack[4]; float* bsum[4];
  for (int i = 0; i < 4; ++i)
    lpack[i] = (unsigned short*)alloc((size_t)NT[i] * KC[i] * 512 * 2);
  for (int i = 0; i < 4; ++i)
    bsum[i] = (float*)alloc((size_t)4 * HS[i] * 4);

  // ---- weight packing ----
  for (int i = 0; i < 6; ++i) {
    int total = mt[i] * nc[i] * 512;
    pack_conv_kernel<<<(total + 255) / 256, 256, 0, stream>>>(
        convw[i], cpack[i], CO[i], CI[i] * KK[i], nc[i], mt[i]);
  }
  for (int i = 0; i < 4; ++i) {
    int total = NT[i] * KC[i] * 512;
    pack_lstm_kernel<<<(total + 255) / 256, 256, 0, stream>>>(
        Wih[i], Whh[i], lpack[i], INS[i], HS[i], NT[i], KC[i]);
    int g = 4 * HS[i];
    bias_sum_kernel<<<(g + 255) / 256, 256, 0, stream>>>(bih[i], bhh[i], bsum[i], g);
  }

  // ---- conv stack (ping-pong activations, bf16) ----
  unsigned short* bufs[2] = {bufA, bufB};
  const unsigned short* cur = nullptr;
  for (int i = 0; i < 6; ++i) {
    int Lin = LINA[i];
    int Lc  = Lin - KK[i] + 1;
    int Lp  = LINA[i + 1];
    dim3 grid((Lc + 15) / 16, 512);
    unsigned short* outp = bufs[i & 1];
    if (i == 0)
      conv_pool_kernel<float><<<grid, 64, 0, stream>>>(
          (const float*)d_in[input_idx], cpack[0], convb[0], outp,
          CI[0], CO[0], KK[0], Lin, Lc, Lp, nc[0], mt[0]);
    else
      conv_pool_kernel<unsigned short><<<grid, 64, 0, stream>>>(
          cur, cpack[i], convb[i], outp,
          CI[i], CO[i], KK[i], Lin, Lc, Lp, nc[i], mt[i]);
    cur = outp;
  }

  // ---- fused LSTM + classifier ----
  lstm_head_kernel<<<8, 128, 0, stream>>>(
      cur, lpack[0], lpack[1], lpack[2], lpack[3],
      bsum[0], bsum[1], bsum[2], bsum[3],
      clsw, clsb, (float*)d_out);
}
